// MultiHeadSelfAttention_15281493639358
// MI455X (gfx1250) — matrix operand order repaired and checked by reading, not compiled, not run on hardware
//
#include <hip/hip_runtime.h>

typedef __bf16 bf16;
typedef __attribute__((ext_vector_type(16))) __bf16 v16bf;
typedef __attribute__((ext_vector_type(8)))  __bf16 v8bf;
typedef __attribute__((ext_vector_type(8)))  float  v8f;

#define DMODEL 512
#define TSEQ   2048
#define BATCH  4
#define NHEAD  8
#define HDIM   64
#define N3     1536   // 3*DMODEL

__device__ __forceinline__ unsigned laneid() { return threadIdx.x & 31u; }

// Intra-wave LDS RAW fence (producer lanes != consumer lanes, same wave).
__device__ __forceinline__ void waitds() { asm volatile("s_wait_dscnt 0" ::: "memory"); }

__device__ __forceinline__ v8f wmma_bf16(v16bf a, v16bf b, v8f c) {
  return __builtin_amdgcn_wmma_f32_16x16x32_bf16(false, a, false, b, (short)0, c,
                                                 false, false);
}

// ---- Fragment loaders (wave32); A and B share contiguous k = 16*(lane/16) + s ----
// A: 16x32 MxK bf16, row-major source. lane row = lane%16;
// K elements: [kg*16 .. kg*16+7] and [kg*16+8 .. kg*16+15]  (kg = lane/16).
// Two contiguous 16-byte loads per lane. Requires 16B alignment of rows.
__device__ __forceinline__ v16bf load_a_bf(const bf16* __restrict__ src, int ld) {
  unsigned lane = laneid(), row = lane & 15u, kg = lane >> 4;
  const bf16* p = src + (size_t)row * ld + kg * 16;
  v8bf lo = *(const v8bf*)(p);
  v8bf hi = *(const v8bf*)(p + 8);
  v16bf a;
#pragma unroll
  for (int e = 0; e < 8; ++e) { a[e] = lo[e]; a[8 + e] = hi[e]; }
  return a;
}

// B: 32x16 KxN where memory holds B^T (element (k,n) at src[n*ld + k]).
// col = lane%16, lanes<16 hold K=0..15, lanes>=16 hold K=16..31.
// One contiguous 32-byte load per lane. Requires 32B alignment.
__device__ __forceinline__ v16bf load_bt_bf(const bf16* __restrict__ src, int ld) {
  unsigned lane = laneid(), col = lane & 15u, kh = lane >> 4;
  return *(const v16bf*)(src + (size_t)col * ld + kh * 16);
}

// ============ Kernel 0: one-time precision/layout prep ============
// x -> bf16 (row-major);  W_qkv -> bf16 transposed [N3, D];  W_out -> bf16 transposed [D, D].
__global__ __launch_bounds__(256) void prep_kernel(
    const float* __restrict__ x, const float* __restrict__ Wqkv,
    const float* __restrict__ Wout, bf16* __restrict__ xb,
    bf16* __restrict__ Wqkv_t, bf16* __restrict__ Wout_t) {
  const size_t stride = (size_t)gridDim.x * 256;
  const size_t i0 = (size_t)blockIdx.x * 256 + threadIdx.x;
  const size_t NX = (size_t)BATCH * TSEQ * DMODEL;
  for (size_t i = i0; i < NX; i += stride) xb[i] = (bf16)x[i];
  const size_t NW = (size_t)DMODEL * N3;
  for (size_t i = i0; i < NW; i += stride) {
    size_t k = i / N3, n = i % N3;
    Wqkv_t[n * DMODEL + k] = (bf16)Wqkv[i];
  }
  const size_t NO = (size_t)DMODEL * DMODEL;
  for (size_t i = i0; i < NO; i += stride) {
    size_t k = i / DMODEL, n = i % DMODEL;
    Wout_t[n * DMODEL + k] = (bf16)Wout[i];
  }
}

// ============ Kernel 1: qkv = x @ W_qkv + b; scatter Q,K [B,H,T,64], V^T [B,H,64,T] ============
__global__ __launch_bounds__(256) void qkv_proj_kernel(
    const bf16* __restrict__ xb, const bf16* __restrict__ Wt,
    const float* __restrict__ bias, bf16* __restrict__ Q,
    bf16* __restrict__ Kd, bf16* __restrict__ Vt) {
  const int w  = blockIdx.x * 8 + (threadIdx.x >> 5);
  const int mt = w / (N3 / 64);
  const int nt = w % (N3 / 64);
  const int m0 = mt * 16, n0 = nt * 64;
  v8f c0 = {}, c1 = {}, c2 = {}, c3 = {};
  for (int k = 0; k < DMODEL; k += 32) {
    v16bf a  = load_a_bf(xb + (size_t)m0 * DMODEL + k, DMODEL);
    v16bf b0 = load_bt_bf(Wt + (size_t)(n0 +  0) * DMODEL + k, DMODEL);
    v16bf b1 = load_bt_bf(Wt + (size_t)(n0 + 16) * DMODEL + k, DMODEL);
    v16bf b2 = load_bt_bf(Wt + (size_t)(n0 + 32) * DMODEL + k, DMODEL);
    v16bf b3 = load_bt_bf(Wt + (size_t)(n0 + 48) * DMODEL + k, DMODEL);
    c0 = wmma_bf16(a, b0, c0);
    c1 = wmma_bf16(a, b1, c1);
    c2 = wmma_bf16(a, b2, c2);
    c3 = wmma_bf16(a, b3, c3);
  }
  v8f cc[4] = {c0, c1, c2, c3};
  unsigned lane = laneid(), col = lane & 15u, rh = lane >> 4;
#pragma unroll
  for (int t = 0; t < 4; ++t) {
    int n   = n0 + t * 16 + (int)col;
    int sec = n / DMODEL;            // 0=Q, 1=K, 2=V (64-wide tile is one section/head)
    int d   = n % DMODEL;
    int h   = d / HDIM, dd = d % HDIM;
    float scale = (sec == 0) ? 0.125f : 1.0f;   // fold 1/sqrt(64) into Q
    float bv    = bias[n];
#pragma unroll
    for (int r = 0; r < 8; ++r) {
      int g  = m0 + r + 8 * (int)rh;            // row in [0, B*T)
      int bb = g / TSEQ, tt = g % TSEQ;
      bf16 val = (bf16)((cc[t][r] + bv) * scale);
      if (sec == 2) {
        Vt[(((size_t)bb * NHEAD + h) * HDIM + dd) * TSEQ + tt] = val;
      } else {
        bf16* dst = (sec == 0) ? Q : Kd;
        dst[(((size_t)bb * NHEAD + h) * TSEQ + tt) * HDIM + dd] = val;
      }
    }
  }
}

// ============ Kernel 2: causal flash attention, one wave per (b,h,16-query tile) ============
__global__ __launch_bounds__(256) void flash_attn_kernel(
    const bf16* __restrict__ Q, const bf16* __restrict__ K,
    const bf16* __restrict__ Vt, bf16* __restrict__ ATT) {
  __shared__ float sS[8][16][32];                 // raw scores, per wave
  __shared__ __align__(32) bf16 sP[8][16][32];    // probabilities (A operand for P*V)
  __shared__ float sRow[8][16];                   // per-row broadcast (alpha / 1/l)

  const int wid = threadIdx.x >> 5;
  const int w   = blockIdx.x * 8 + wid;
  const int qt  = w & 127;                 // T/16 = 128 query tiles
  const int bh  = w >> 7;
  const int b   = bh >> 3, h = bh & 7;
  const int m0  = qt * 16;
  const size_t hoff = ((size_t)b * NHEAD + h) * TSEQ * HDIM;
  const bf16* Qh  = Q  + hoff;
  const bf16* Kh  = K  + hoff;
  const bf16* Vth = Vt + hoff;             // [64, T] per head

  v16bf qa0 = load_a_bf(Qh + (size_t)m0 * HDIM + 0,  HDIM);   // d = 0..31
  v16bf qa1 = load_a_bf(Qh + (size_t)m0 * HDIM + 32, HDIM);   // d = 32..63

  v8f o0 = {}, o1 = {}, o2 = {}, o3 = {};
  float m_i = -1e30f, l_i = 0.0f;          // owned by lanes 0..15 (row = lane)
  unsigned lane = laneid(), col = lane & 15u, rh = lane >> 4;

  const int ktmax = (m0 + 15) >> 5;        // last 32-key tile touching the diagonal
  for (int kt = 0; kt <= ktmax; ++kt) {
    const int k0 = kt * 32;
    // --- S = Q K^T (16x32 tile = two 16x16 C tiles, K-dim = 64 in two chunks) ---
    v8f s0 = {}, s1 = {};
    {
      v16bf bk;
      bk = load_bt_bf(Kh + (size_t)(k0 +  0) * HDIM +  0, HDIM); s0 = wmma_bf16(qa0, bk, s0);
      bk = load_bt_bf(Kh + (size_t)(k0 + 16) * HDIM +  0, HDIM); s1 = wmma_bf16(qa0, bk, s1);
      bk = load_bt_bf(Kh + (size_t)(k0 +  0) * HDIM + 32, HDIM); s0 = wmma_bf16(qa1, bk, s0);
      bk = load_bt_bf(Kh + (size_t)(k0 + 16) * HDIM + 32, HDIM); s1 = wmma_bf16(qa1, bk, s1);
    }
    // --- stage scores to LDS (C layout -> row-major) ---
#pragma unroll
    for (int r = 0; r < 8; ++r) {
      sS[wid][r + 8 * rh][col]      = s0[r];
      sS[wid][r + 8 * rh][16 + col] = s1[r];
    }
    waitds();
    // --- online softmax: lane r < 16 owns query row m0+r ---
    if (lane < 16) {
      const int r = (int)lane;
      const int kmaxv = (m0 + r) - k0;     // valid keys: local idx <= kmaxv (>=0 always)
      float tmax = -1e30f;
#pragma unroll
      for (int c = 0; c < 32; ++c) {
        float v2 = (c <= kmaxv) ? sS[wid][r][c] : -1e30f;
        tmax = fmaxf(tmax, v2);
      }
      float mnew  = fmaxf(m_i, tmax);
      float alpha = __expf(m_i - mnew);
      float lsum  = 0.0f;
#pragma unroll
      for (int c = 0; c < 32; ++c) {
        float p = (c <= kmaxv) ? __expf(sS[wid][r][c] - mnew) : 0.0f;
        sP[wid][r][c] = (bf16)p;
        lsum += p;
      }
      l_i = l_i * alpha + lsum;
      m_i = mnew;
      sRow[wid][r] = alpha;
    }
    waitds();
    // --- rescale O accumulators by per-row alpha ---
    float al[8];
#pragma unroll
    for (int r = 0; r < 8; ++r) al[r] = sRow[wid][r + 8 * rh];
#pragma unroll
    for (int r = 0; r < 8; ++r) {
      o0[r] *= al[r]; o1[r] *= al[r]; o2[r] *= al[r]; o3[r] *= al[r];
    }
    // --- O += P V (P: 16x32 A fragment from LDS; V^T gives contiguous B fragments) ---
    v16bf pa  = load_a_bf(&sP[wid][0][0], 32);
    v16bf bv0 = load_bt_bf(Vth + (size_t) 0 * TSEQ + k0, TSEQ);
    v16bf bv1 = load_bt_bf(Vth + (size_t)16 * TSEQ + k0, TSEQ);
    v16bf bv2 = load_bt_bf(Vth + (size_t)32 * TSEQ + k0, TSEQ);
    v16bf bv3 = load_bt_bf(Vth + (size_t)48 * TSEQ + k0, TSEQ);
    o0 = wmma_bf16(pa, bv0, o0);
    o1 = wmma_bf16(pa, bv1, o1);
    o2 = wmma_bf16(pa, bv2, o2);
    o3 = wmma_bf16(pa, bv3, o3);
  }
  // --- normalize and write [B,T,D] bf16 ---
  if (lane < 16) sRow[wid][lane] = 1.0f / l_i;
  waitds();
  float il[8];
#pragma unroll
  for (int r = 0; r < 8; ++r) il[r] = sRow[wid][r + 8 * rh];
  v8f oo[4] = {o0, o1, o2, o3};
#pragma unroll
  for (int t = 0; t < 4; ++t) {
#pragma unroll
    for (int r = 0; r < 8; ++r) {
      int g = b * TSEQ + m0 + r + 8 * (int)rh;
      ATT[(size_t)g * DMODEL + h * HDIM + t * 16 + col] = (bf16)(oo[t][r] * il[r]);
    }
  }
}

// ============ Kernel 3: out = attn @ W_out + b_out (f32 output) ============
__global__ __launch_bounds__(256) void out_proj_kernel(
    const bf16* __restrict__ A, const bf16* __restrict__ Wt,
    const float* __restrict__ bias, float* __restrict__ out) {
  const int w  = blockIdx.x * 8 + (threadIdx.x >> 5);
  const int mt = w / (DMODEL / 64);
  const int nt = w % (DMODEL / 64);
  const int m0 = mt * 16, n0 = nt * 64;
  v8f c0 = {}, c1 = {}, c2 = {}, c3 = {};
  for (int k = 0; k < DMODEL; k += 32) {
    v16bf a  = load_a_bf(A + (size_t)m0 * DMODEL + k, DMODEL);
    v16bf b0 = load_bt_bf(Wt + (size_t)(n0 +  0) * DMODEL + k, DMODEL);
    v16bf b1 = load_bt_bf(Wt + (size_t)(n0 + 16) * DMODEL + k, DMODEL);
    v16bf b2 = load_bt_bf(Wt + (size_t)(n0 + 32) * DMODEL + k, DMODEL);
    v16bf b3 = load_bt_bf(Wt + (size_t)(n0 + 48) * DMODEL + k, DMODEL);
    c0 = wmma_bf16(a, b0, c0);
    c1 = wmma_bf16(a, b1, c1);
    c2 = wmma_bf16(a, b2, c2);
    c3 = wmma_bf16(a, b3, c3);
  }
  v8f cc[4] = {c0, c1, c2, c3};
  unsigned lane = laneid(), col = lane & 15u, rh = lane >> 4;
#pragma unroll
  for (int t = 0; t < 4; ++t) {
    int n = n0 + t * 16 + (int)col;
    float bv = bias[n];
#pragma unroll
    for (int r = 0; r < 8; ++r) {
      int g = m0 + r + 8 * (int)rh;
      out[(size_t)g * DMODEL + n] = cc[t][r] + bv;
    }
  }
}

extern "C" void kernel_launch(void* const* d_in, const int* in_sizes, int n_in,
                              void* d_out, int out_size, void* d_ws, size_t ws_size,
                              hipStream_t stream) {
  (void)in_sizes; (void)n_in; (void)out_size; (void)ws_size;
  const float* x    = (const float*)d_in[0];   // [4,2048,512]
  const float* Wqkv = (const float*)d_in[1];   // [512,1536]
  const float* bqkv = (const float*)d_in[2];   // [1536]
  const float* Wout = (const float*)d_in[3];   // [512,512]
  const float* bout = (const float*)d_in[4];   // [512]
  float* out = (float*)d_out;                  // [4,2048,512] f32

  // Workspace layout (bf16 elements), all regions 64B-aligned by construction:
  //   Q   [B,H,T,64]    4M elems
  //   K   [B,H,T,64]    4M elems
  //   Vt  [B,H,64,T]    4M elems
  //   ATT [B,T,D]       4M elems
  //   xb  [B*T, D]      4M elems
  //   Wqkv_t [N3, D]    768K elems
  //   Wout_t [D, D]     256K elems         total ~ 42 MB
  const size_t per = (size_t)BATCH * NHEAD * TSEQ * HDIM;   // 4M
  bf16* Q      = (bf16*)d_ws;
  bf16* K      = Q + per;
  bf16* Vt     = K + per;
  bf16* ATT    = Vt + per;
  bf16* xb     = ATT + per;
  bf16* Wqkv_t = xb + per;
  bf16* Wout_t = Wqkv_t + (size_t)N3 * DMODEL;

  prep_kernel<<<dim3(1024), dim3(256), 0, stream>>>(x, Wqkv, Wout, xb, Wqkv_t, Wout_t);
  // 8192x1536 GEMM: 512 m-tiles * 24 n-tiles(64-wide) = 12288 waves / 8 per block.
  qkv_proj_kernel<<<dim3(1536), dim3(256), 0, stream>>>(xb, Wqkv_t, bqkv, Q, K, Vt);
  // 4*8*128 = 4096 waves / 8 per block.
  flash_attn_kernel<<<dim3(512), dim3(256), 0, stream>>>(Q, K, Vt, ATT);
  // 8192x512 GEMM: 512 * 8 = 4096 waves / 8 per block.
  out_proj_kernel<<<dim3(512), dim3(256), 0, stream>>>(ATT, Wout_t, bout, out);
}
